// PointDecoder_36283883716863
// MI455X (gfx1250) — matrix operand order repaired and checked by reading, not compiled, not run on hardware
//
#include <hip/hip_runtime.h>
#include <math.h>
#include <stdint.h>

// ---------------------------------------------------------------------------
// Types for WMMA
// ---------------------------------------------------------------------------
typedef __attribute__((ext_vector_type(16))) _Float16 v16h;
typedef __attribute__((ext_vector_type(8)))  float    v8f;

// Address-space-qualified pointer types for the async global->LDS builtin.
typedef __attribute__((address_space(1))) int as1_int;
typedef __attribute__((address_space(3))) int as3_int;

// Probe for CDNA5 async global->LDS copy builtins (guarded; plain fallback).
#if defined(__has_builtin)
#if __has_builtin(__builtin_amdgcn_global_load_async_to_lds_b32) && \
    __has_builtin(__builtin_amdgcn_s_wait_asynccnt)
#define HAVE_ASYNC_LDS 1
#endif
#endif

// ---------------------------------------------------------------------------
// WMMA GEMM: C[M,N] = (A [+pe broadcast]) @ W + bias
// A row-major f32 [M,K], W row-major f32 [K,N]. M%64==0, N%32==0, K%32==0, K<=256.
// Block = 4 waves; block owns a 32-column slice of W staged once in LDS
// (stride 34 floats -> WMMA B-fragment lane halves hit disjoint bank groups).
// Each wave computes two 16x16 C tiles sharing one A fragment per K-step.
// ---------------------------------------------------------------------------
template <int ADD_PE>
__global__ __launch_bounds__(128) void wmma_gemm_kernel(
    const float* __restrict__ A, const float* __restrict__ pe, int peMask,
    const float* __restrict__ W, const float* __restrict__ bias,
    float* __restrict__ C, int M, int K, int N) {
  constexpr int BSTR = 34;                  // LDS row stride (floats)
  __shared__ float Bs[256 * BSTR];          // up to K=256 x 32 cols (padded)

  const int tid  = threadIdx.x;
  const int lane = tid & 31;
  const int wave = tid >> 5;
  const int n0b  = blockIdx.x * 32;         // block's column base
  const int m0   = (blockIdx.y * 4 + wave) * 16;
  const int half = (lane >> 4) & 1;         // 0: lanes 0-15, 1: lanes 16-31
  const int l16  = lane & 15;
  const int arow = m0 + l16;
  (void)M;

  // ---- stage W[:, n0b : n0b+32] into LDS (once per block) ----
  for (int idx = tid; idx < K * 32; idx += 128) {
    int k = idx >> 5, c = idx & 31;
#if defined(HAVE_ASYNC_LDS)
    __builtin_amdgcn_global_load_async_to_lds_b32(
        (as1_int*)(uintptr_t)(W + (size_t)k * N + n0b + c),
        (as3_int*)&Bs[k * BSTR + c], 0, 0);
#else
    Bs[k * BSTR + c] = W[(size_t)k * N + n0b + c];
#endif
  }
#if defined(HAVE_ASYNC_LDS)
  __builtin_amdgcn_s_wait_asynccnt(0);
#endif
  __syncthreads();

  const float* Arow = A + (size_t)arow * K;
  const float* Prow = ADD_PE ? (pe + (size_t)(arow & peMask) * K) : nullptr;

  v8f acc0 = {};
  v8f acc1 = {};
  for (int kt = 0; kt < K; kt += 32) {
    if (kt + 32 < K) __builtin_prefetch(Arow + kt + 32, 0, 3);
    v16h av, bv0, bv1;
    // A fragment: 16x32 f16. VGPR v (v<4): K = 2(v&3)+8*half ; (v>=4): +16.
#pragma unroll
    for (int v = 0; v < 8; ++v) {
      int k0 = kt + ((v & 4) ? 16 : 0) + ((v & 3) << 1) + (half ? 8 : 0);
      float x0 = Arow[k0];
      float x1 = Arow[k0 + 1];
      if (ADD_PE) { x0 += Prow[k0]; x1 += Prow[k0 + 1]; }
      av[2 * v]     = (_Float16)x0;
      av[2 * v + 1] = (_Float16)x1;
    }
    // B fragments from LDS: VGPR v (v<4): K = 2(v&3)+8*half ; (v>=4): +16, N = tile base + (lane&15)
#pragma unroll
    for (int v = 0; v < 8; ++v) {
      int kg = kt + ((v & 4) ? 16 : 0) + ((v & 3) << 1) + (half ? 8 : 0);
      bv0[2 * v]     = (_Float16)Bs[kg * BSTR + l16];
      bv0[2 * v + 1] = (_Float16)Bs[(kg + 1) * BSTR + l16];
      bv1[2 * v]     = (_Float16)Bs[kg * BSTR + 16 + l16];
      bv1[2 * v + 1] = (_Float16)Bs[(kg + 1) * BSTR + 16 + l16];
    }
    acc0 = __builtin_amdgcn_wmma_f32_16x16x32_f16(
        false, av, false, bv0, (short)0, acc0, false, false);
    acc1 = __builtin_amdgcn_wmma_f32_16x16x32_f16(
        false, av, false, bv1, (short)0, acc1, false, false);
  }

  float bia0 = bias ? bias[n0b + l16] : 0.0f;
  float bia1 = bias ? bias[n0b + 16 + l16] : 0.0f;
  // C/D layout: VGPR r -> row m0 + r + 8*half, col base + (lane&15)
#pragma unroll
  for (int r = 0; r < 8; ++r) {
    int row = m0 + r + half * 8;
    C[(size_t)row * N + n0b + l16]      = acc0[r] + bia0;
    C[(size_t)row * N + n0b + 16 + l16] = acc1[r] + bia1;
  }
}

// ---------------------------------------------------------------------------
// Positional encoding: pe[4096, 256]
// ---------------------------------------------------------------------------
__global__ __launch_bounds__(128) void pe_kernel(const float* __restrict__ G,
                                                 float* __restrict__ pe) {
  int pos = blockIdx.x;           // y*64+x
  int j = threadIdx.x;            // 0..127
  int y = pos >> 6, x = pos & 63;
  float X = 2.0f * ((x + 0.5f) / 64.0f) - 1.0f;
  float Y = 2.0f * ((y + 0.5f) / 64.0f) - 1.0f;
  float t = 6.283185307179586f * (X * G[j] + Y * G[128 + j]);
  pe[(size_t)pos * 256 + j]       = sinf(t);
  pe[(size_t)pos * 256 + 128 + j] = cosf(t);
}

// ---------------------------------------------------------------------------
// [B,C,HW] -> [B,HW,C] tiled transpose (keys init)
// ---------------------------------------------------------------------------
__global__ __launch_bounds__(256) void transpose_kernel(
    const float* __restrict__ img, float* __restrict__ keys) {
  __shared__ float tile[32][33];
  int b  = blockIdx.z;
  int n0 = blockIdx.x * 32;
  int c0 = blockIdx.y * 32;
  int tx = threadIdx.x, ty = threadIdx.y;   // 32 x 8
#pragma unroll
  for (int k = 0; k < 4; ++k) {
    int c = c0 + ty + k * 8;
    tile[ty + k * 8][tx] = img[((size_t)b * 256 + c) * 4096 + n0 + tx];
  }
  __syncthreads();
#pragma unroll
  for (int k = 0; k < 4; ++k) {
    int n = n0 + ty + k * 8;
    keys[((size_t)b * 4096 + n) * 256 + c0 + tx] = tile[tx][ty + k * 8];
  }
}

__global__ void init_queries_kernel(const float* __restrict__ mask,
                                    float* __restrict__ q) {
  q[blockIdx.x * 256 + threadIdx.x] = mask[threadIdx.x];
}

// ---------------------------------------------------------------------------
// Small dense layer on the (few) query tokens: Y = act(Xeff @ W + b) + res
// Xeff = X (+ addVec broadcast). act: 0=none, 1=relu.
// ---------------------------------------------------------------------------
__global__ void small_linear_kernel(const float* __restrict__ X,
                                    const float* __restrict__ addVec,
                                    const float* __restrict__ W,
                                    const float* __restrict__ bias,
                                    const float* __restrict__ res,
                                    float* __restrict__ Y,
                                    int din, int dout, int act) {
  int b = blockIdx.x;
  int j = blockIdx.y * blockDim.x + threadIdx.x;
  if (j >= dout) return;
  const float* x = X + (size_t)b * din;
  float s = bias ? bias[j] : 0.0f;
  for (int k = 0; k < din; ++k) {
    float xv = x[k];
    if (addVec) xv += addVec[k];
    s += xv * W[(size_t)k * dout + j];
  }
  if (act == 1) s = fmaxf(s, 0.0f);
  if (res) s += res[(size_t)b * dout + j];
  Y[(size_t)b * dout + j] = s;
}

// ---------------------------------------------------------------------------
// Row LayerNorm over 256 elements, in place (eps 1e-5)
// ---------------------------------------------------------------------------
__global__ __launch_bounds__(256) void ln_rows_kernel(float* __restrict__ X,
                                                      const float* __restrict__ w,
                                                      const float* __restrict__ b) {
  __shared__ float red[256];
  int t = threadIdx.x;
  float* x = X + (size_t)blockIdx.x * 256;
  float v = x[t];
  red[t] = v; __syncthreads();
  for (int s = 128; s > 0; s >>= 1) { if (t < s) red[t] += red[t + s]; __syncthreads(); }
  float mean = red[0] * (1.0f / 256.0f); __syncthreads();
  float d = v - mean;
  red[t] = d * d; __syncthreads();
  for (int s = 128; s > 0; s >>= 1) { if (t < s) red[t] += red[t + s]; __syncthreads(); }
  float var = red[0] * (1.0f / 256.0f);
  x[t] = d * rsqrtf(var + 1e-5f) * w[t] + b[t];
}

// keys = LN(keys + broadcast ov[b,:]) with norm weights, in place
__global__ __launch_bounds__(256) void keys_add_ln_kernel(
    float* __restrict__ keys, const float* __restrict__ ov,
    const float* __restrict__ w, const float* __restrict__ b) {
  __shared__ float red[256];
  size_t r = blockIdx.x;            // b*4096 + n
  int bb = (int)(r >> 12);
  int t = threadIdx.x;
  float* x = keys + r * 256;
  float v = x[t] + ov[bb * 256 + t];
  red[t] = v; __syncthreads();
  for (int s = 128; s > 0; s >>= 1) { if (t < s) red[t] += red[t + s]; __syncthreads(); }
  float mean = red[0] * (1.0f / 256.0f); __syncthreads();
  float d = v - mean;
  red[t] = d * d; __syncthreads();
  for (int s = 128; s > 0; s >>= 1) { if (t < s) red[t] += red[t + s]; __syncthreads(); }
  float var = red[0] * (1.0f / 256.0f);
  x[t] = d * rsqrtf(var + 1e-5f) * w[t] + b[t];
}

// ---------------------------------------------------------------------------
// Token-to-image attention: 1 query token, 4096 keys, 8 heads x ch16.
// One block per (b, head). Deterministic tree reductions.
// ---------------------------------------------------------------------------
__global__ __launch_bounds__(256) void attn1q_kernel(
    const float* __restrict__ q, const float* __restrict__ kk,
    const float* __restrict__ vv, float* __restrict__ out) {
  __shared__ float sc[4096];
  __shared__ float red[256];
  __shared__ float osum[16];
  int b = blockIdx.x >> 3, h = blockIdx.x & 7;
  int t = threadIdx.x;
  const float* qh = q + b * 128 + h * 16;
  float qv[16];
#pragma unroll
  for (int c = 0; c < 16; ++c) qv[c] = qh[c];
  const float* kb = kk + (size_t)b * 4096 * 128 + h * 16;
  float lmax = -3.0e38f;
  for (int n = t; n < 4096; n += 256) {
    const float* kr = kb + (size_t)n * 128;
    float s = 0.0f;
#pragma unroll
    for (int c = 0; c < 16; ++c) s += qv[c] * kr[c];
    s *= 0.25f;                       // 1/sqrt(16)
    sc[n] = s;
    lmax = fmaxf(lmax, s);
  }
  red[t] = lmax; __syncthreads();
  for (int s = 128; s > 0; s >>= 1) { if (t < s) red[t] = fmaxf(red[t], red[t + s]); __syncthreads(); }
  float mx = red[0]; __syncthreads();
  float lsum = 0.0f;
  for (int n = t; n < 4096; n += 256) {
    float e = expf(sc[n] - mx);
    sc[n] = e;
    lsum += e;
  }
  red[t] = lsum; __syncthreads();
  for (int s = 128; s > 0; s >>= 1) { if (t < s) red[t] += red[t + s]; __syncthreads(); }
  float inv = 1.0f / red[0]; __syncthreads();

  const float* vb = vv + (size_t)b * 4096 * 128 + h * 16;
  float po[16];
#pragma unroll
  for (int c = 0; c < 16; ++c) po[c] = 0.0f;
  for (int n = t; n < 4096; n += 256) {
    float a = sc[n];
    const float* vr = vb + (size_t)n * 128;
#pragma unroll
    for (int c = 0; c < 16; ++c) po[c] += a * vr[c];
  }
  // deterministic per-channel tree reduction
  for (int c = 0; c < 16; ++c) {
    red[t] = po[c]; __syncthreads();
    for (int s = 128; s > 0; s >>= 1) { if (t < s) red[t] += red[t + s]; __syncthreads(); }
    if (t == 0) osum[c] = red[0] * inv;
    __syncthreads();
  }
  if (t < 16) out[b * 128 + h * 16 + t] = osum[t];
}

// ---------------------------------------------------------------------------
// up1 scatter + bias + LayerNorm2d(eps 1e-6) + exact GELU -> u1 [B,16384,64]
// ---------------------------------------------------------------------------
__global__ __launch_bounds__(64) void up1_assemble_kernel(
    const float* __restrict__ Y1, const float* __restrict__ bias,
    const float* __restrict__ lw, const float* __restrict__ lb,
    float* __restrict__ u1) {
  __shared__ float red[64];
  int pid = blockIdx.x;             // b*16384 + oy*128 + ox
  int b = pid >> 14;
  int p2 = pid & 16383;
  int oy = p2 >> 7, ox = p2 & 127;
  int q = ((oy & 1) << 1) | (ox & 1);
  size_t srow = (size_t)b * 4096 + (size_t)(oy >> 1) * 64 + (ox >> 1);
  int o = threadIdx.x;
  float v = Y1[srow * 256 + o * 4 + q] + bias[o];
  red[o] = v; __syncthreads();
  for (int s = 32; s > 0; s >>= 1) { if (o < s) red[o] += red[o + s]; __syncthreads(); }
  float mean = red[0] * (1.0f / 64.0f); __syncthreads();
  float d = v - mean;
  red[o] = d * d; __syncthreads();
  for (int s = 32; s > 0; s >>= 1) { if (o < s) red[o] += red[o + s]; __syncthreads(); }
  float var = red[0] * (1.0f / 64.0f);
  float y = d * rsqrtf(var + 1e-6f) * lw[o] + lb[o];
  y = 0.5f * y * (1.0f + erff(y * 0.7071067811865476f));   // exact GELU
  u1[(size_t)pid * 64 + o] = y;
}

// ---------------------------------------------------------------------------
// heat[b,256,256] = sum_o hyper[b,o] * gelu(Y2[row2, o*4+q] + b2[o])
// ---------------------------------------------------------------------------
__global__ __launch_bounds__(256) void heat_kernel(
    const float* __restrict__ Y2, const float* __restrict__ b2,
    const float* __restrict__ hyp, float* __restrict__ heat) {
  int b = blockIdx.x;
  int pos = blockIdx.y * 256 + threadIdx.x;   // 0..65535
  int oy = pos >> 8, ox = pos & 255;
  int q = ((oy & 1) << 1) | (ox & 1);
  size_t row2 = (size_t)b * 16384 + (size_t)(oy >> 1) * 128 + (ox >> 1);
  const float* yr = Y2 + row2 * 128;
  const float* hb = hyp + b * 32;
  float acc = 0.0f;
#pragma unroll
  for (int o = 0; o < 32; ++o) {
    float v = yr[o * 4 + q] + b2[o];
    v = 0.5f * v * (1.0f + erff(v * 0.7071067811865476f));
    acc += hb[o] * v;
  }
  heat[(size_t)b * 65536 + pos] = acc;
}

// ---------------------------------------------------------------------------
// 3x3 NMS (SAME padding, OOB excluded)
// ---------------------------------------------------------------------------
__global__ __launch_bounds__(256) void nms_kernel(const float* __restrict__ heat,
                                                  float* __restrict__ nms) {
  int b = blockIdx.x;
  int pos = blockIdx.y * 256 + threadIdx.x;
  int y = pos >> 8, x = pos & 255;
  const float* hb = heat + (size_t)b * 65536;
  float c = hb[pos];
  float m = -3.0e38f;
  for (int dy = -1; dy <= 1; ++dy)
    for (int dx = -1; dx <= 1; ++dx) {
      int ny = y + dy, nx = x + dx;
      if (ny >= 0 && ny < 256 && nx >= 0 && nx < 256)
        m = fmaxf(m, hb[ny * 256 + nx]);
    }
  nms[(size_t)b * 65536 + pos] = (m == c) ? c : 0.0f;
}

// ---------------------------------------------------------------------------
// Deterministic top-1000 with threshold; ties broken by smaller index
// (matches jax.lax.top_k ordering). One block per batch.
// ---------------------------------------------------------------------------
__global__ __launch_bounds__(1024) void topk_kernel(
    const float* __restrict__ nms, float* __restrict__ mvals,
    float* __restrict__ pts, float* __restrict__ scores) {
  __shared__ float rv[1024];
  __shared__ int   ri[1024];
  int b = blockIdx.x;
  int t = threadIdx.x;
  const float* src = nms + (size_t)b * 65536;
  float* mv = mvals + (size_t)b * 65536;
  for (int i = t; i < 65536; i += 1024) {
    float v = src[i];
    mv[i] = (v > 0.1f) ? v : -3.0e38f;
  }
  __syncthreads();
  for (int sel = 0; sel < 1000; ++sel) {
    float bv = -3.0e38f; int bi = 0x7fffffff;
    for (int i = t; i < 65536; i += 1024) {
      float v = mv[i];
      if (v > bv || (v == bv && i < bi)) { bv = v; bi = i; }
    }
    rv[t] = bv; ri[t] = bi; __syncthreads();
    for (int s = 512; s > 0; s >>= 1) {
      if (t < s) {
        float v2 = rv[t + s]; int i2 = ri[t + s];
        if (v2 > rv[t] || (v2 == rv[t] && i2 < ri[t])) { rv[t] = v2; ri[t] = i2; }
      }
      __syncthreads();
    }
    if (t == 0) {
      float best = rv[0]; int bidx = ri[0];
      bool valid = best > 0.1f;
      scores[(size_t)b * 1000 + sel] = valid ? best : 0.0f;
      pts[((size_t)b * 1000 + sel) * 2 + 0] = valid ? (float)(bidx & 255) * 4.0f : 0.0f;
      pts[((size_t)b * 1000 + sel) * 2 + 1] = valid ? (float)(bidx >> 8) * 4.0f : 0.0f;
      if (valid) mv[bidx] = -3.0e38f;
    }
    __syncthreads();
  }
}

// ---------------------------------------------------------------------------
// Host orchestration
// ---------------------------------------------------------------------------
struct AttnP { const float *qw,*qb,*kw,*kb,*vw,*vb,*ow,*ob; };

extern "C" void kernel_launch(void* const* d_in, const int* in_sizes, int n_in,
                              void* d_out, int out_size, void* d_ws, size_t ws_size,
                              hipStream_t stream) {
  (void)in_sizes; (void)n_in; (void)out_size; (void)ws_size;
  const float* img = (const float*)d_in[0];
  int pi = 1;
  auto NP = [&]() { return (const float*)d_in[pi++]; };
  auto readAttn = [&]() { AttnP a; a.qw=NP();a.qb=NP();a.kw=NP();a.kb=NP();
                          a.vw=NP();a.vb=NP();a.ow=NP();a.ob=NP(); return a; };
  struct LayerP {
    AttnP sa, t2i, i2t;
    const float *n1w,*n1b,*n2w,*n2b,*n3w,*n3b,*n4w,*n4b;
    const float *m1w,*m1b,*m2w,*m2b;
  } L[2];
  for (int i = 0; i < 2; ++i) {
    L[i].sa = readAttn(); L[i].t2i = readAttn(); L[i].i2t = readAttn();
    L[i].n1w=NP();L[i].n1b=NP();L[i].n2w=NP();L[i].n2b=NP();
    L[i].n3w=NP();L[i].n3b=NP();L[i].n4w=NP();L[i].n4b=NP();
    L[i].m1w=NP();L[i].m1b=NP();L[i].m2w=NP();L[i].m2b=NP();
  }
  AttnP FA = readAttn();
  const float* nfw = NP(); const float* nfb = NP();
  const float* mask_tok = NP();
  const float* up1w = NP(); const float* up1b = NP();
  const float* uplnw = NP(); const float* uplnb = NP();
  const float* up2w = NP(); const float* up2b = NP();
  const float* h1w = NP(); const float* h1b = NP();
  const float* h2w = NP(); const float* h2b = NP();
  const float* h3w = NP(); const float* h3b = NP();
  const float* peG = NP();

  // -------- workspace carve (floats) --------
  float* ws = (float*)d_ws;
  size_t off = 0;
  auto alloc = [&](size_t n) { float* p = ws + off; off += n; return p; };
  float* pe   = alloc((size_t)4096 * 256);        //   4 MB
  float* keys = alloc((size_t)16 * 4096 * 256);   //  67 MB
  float* kk   = alloc((size_t)16 * 4096 * 128);   //  33.5 MB
  float* vvb  = alloc((size_t)16 * 4096 * 128);   //  33.5 MB
  float* Y1   = kk;   // alias kk+vv (contiguous 67 MB), free after final attn
  float* u1   = alloc((size_t)16 * 16384 * 64);   //  67 MB
  float* Y2   = alloc((size_t)16 * 16384 * 128);  // 134 MB
  float* mvals = u1;  // reuse after up2 GEMM for top-k scratch
  float* q0 = alloc(16 * 256);
  float* q1 = alloc(16 * 256);
  float* tA = alloc(16 * 2048);   // MLP hidden / generic tmp
  float* tB = alloc(16 * 256);
  float* qproj = alloc(16 * 128);
  float* attno = alloc(16 * 128);
  float* hyper_in = alloc(16 * 32);

  float* outF   = (float*)d_out;
  float* heat   = outF;                                 // 16*65536
  float* pts    = outF + (size_t)16 * 65536;            // 16*1000*2
  float* scores = pts + (size_t)16 * 1000 * 2;          // 16*1000
  float* nmsOut = scores + (size_t)16 * 1000;           // 16*65536

  auto gemm = [&](const float* A, const float* peP, int peMask, const float* W,
                  const float* bias, float* Cc, int M, int K, int N) {
    dim3 g(N / 32, M / 64);
    if (peP)
      wmma_gemm_kernel<1><<<g, 128, 0, stream>>>(A, peP, peMask, W, bias, Cc, M, K, N);
    else
      wmma_gemm_kernel<0><<<g, 128, 0, stream>>>(A, nullptr, 0, W, bias, Cc, M, K, N);
  };
  auto slin = [&](const float* X, const float* addV, const float* W,
                  const float* bias, const float* res, float* Y,
                  int din, int dout, int act) {
    dim3 g(16, (dout + 255) / 256);
    small_linear_kernel<<<g, 256, 0, stream>>>(X, addV, W, bias, res, Y, din, dout, act);
  };

  // -------- prologue --------
  pe_kernel<<<4096, 128, 0, stream>>>(peG, pe);
  transpose_kernel<<<dim3(128, 8, 16), dim3(32, 8), 0, stream>>>(img, keys);
  init_queries_kernel<<<16, 256, 0, stream>>>(mask_tok, q0);

  // -------- 2 two-way transformer layers --------
  for (int i = 0; i < 2; ++i) {
    const LayerP& P = L[i];
    // self-attn on single token: softmax over 1 key == identity on v-path
    slin(q0, nullptr, P.sa.vw, P.sa.vb, nullptr, tA, 256, 256, 0);
    slin(tA, nullptr, P.sa.ow, P.sa.ob, (i == 0 ? nullptr : q0), q1, 256, 256, 0);
    { float* t = q0; q0 = q1; q1 = t; }
    ln_rows_kernel<<<16, 256, 0, stream>>>(q0, P.n1w, P.n1b);

    // t2i cross attention (queries attend image)
    slin(q0, mask_tok, P.t2i.qw, P.t2i.qb, nullptr, qproj, 256, 128, 0);
    gemm(keys, pe, 4095, P.t2i.kw, P.t2i.kb, kk, 65536, 256, 128);    // k = keys+pe
    gemm(keys, nullptr, 0, P.t2i.vw, P.t2i.vb, vvb, 65536, 256, 128); // v = keys
    attn1q_kernel<<<128, 256, 0, stream>>>(qproj, kk, vvb, attno);
    slin(attno, nullptr, P.t2i.ow, P.t2i.ob, q0, q1, 128, 256, 0);
    { float* t = q0; q0 = q1; q1 = t; }
    ln_rows_kernel<<<16, 256, 0, stream>>>(q0, P.n2w, P.n2b);

    // MLP (single token)
    slin(q0, nullptr, P.m1w, P.m1b, nullptr, tA, 256, 2048, 1);
    slin(tA, nullptr, P.m2w, P.m2b, q0, q1, 2048, 256, 0);
    { float* t = q0; q0 = q1; q1 = t; }
    ln_rows_kernel<<<16, 256, 0, stream>>>(q0, P.n3w, P.n3b);

    // i2t: Nk==1 -> softmax==1 -> broadcast (v_proj(queries) @ Wo + bo)
    slin(q0, nullptr, P.i2t.vw, P.i2t.vb, nullptr, qproj, 256, 128, 0);
    slin(qproj, nullptr, P.i2t.ow, P.i2t.ob, nullptr, tB, 128, 256, 0);
    keys_add_ln_kernel<<<65536, 256, 0, stream>>>(keys, tB, P.n4w, P.n4b);
  }

  // -------- final attention + norm --------
  slin(q0, mask_tok, FA.qw, FA.qb, nullptr, qproj, 256, 128, 0);
  gemm(keys, pe, 4095, FA.kw, FA.kb, kk, 65536, 256, 128);
  gemm(keys, nullptr, 0, FA.vw, FA.vb, vvb, 65536, 256, 128);
  attn1q_kernel<<<128, 256, 0, stream>>>(qproj, kk, vvb, attno);
  slin(attno, nullptr, FA.ow, FA.ob, q0, q1, 128, 256, 0);
  { float* t = q0; q0 = q1; q1 = t; }
  ln_rows_kernel<<<16, 256, 0, stream>>>(q0, nfw, nfb);   // q0 = hs

  // -------- upsampling path (ConvT2x2s2 as GEMM) --------
  // up1: [B*4096,256] @ [256, 64*4] ; up1.w torch layout is already row-major [256,256]
  gemm(keys, nullptr, 0, up1w, nullptr, Y1, 65536, 256, 256);
  up1_assemble_kernel<<<262144, 64, 0, stream>>>(Y1, up1b, uplnw, uplnb, u1);
  // up2: [B*16384,64] @ [64, 32*4]
  gemm(u1, nullptr, 0, up2w, nullptr, Y2, 262144, 64, 128);

  // hypernetwork on hs
  slin(q0, nullptr, h1w, h1b, nullptr, tB, 256, 256, 1);
  slin(tB, nullptr, h2w, h2b, nullptr, q1, 256, 256, 1);
  slin(q1, nullptr, h3w, h3b, nullptr, hyper_in, 256, 32, 0);

  // heat + NMS + top-k
  heat_kernel<<<dim3(16, 256), 256, 0, stream>>>(Y2, up2b, hyper_in, heat);
  nms_kernel<<<dim3(16, 256), 256, 0, stream>>>(heat, nmsOut);
  topk_kernel<<<16, 1024, 0, stream>>>(nmsOut, mvals, pts, scores);
}
